// Attn_42408507080957
// MI455X (gfx1250) — compile-verified, matrix operand order checked by reading, not run on hardware
//
#include <hip/hip_runtime.h>
#include <hip/hip_bf16.h>
#include <stdint.h>

// ---------------------------------------------------------------------------
// MI455X / gfx1250, wave32. All GEMMs use v_wmma_f32_16x16x32_bf16.
// B operands pre-swizzled into WMMA fragment order -> contiguous LDS frag loads.
// Q/K/V tiles fetched with the Tensor Data Mover (contiguous 32KB descriptors).
// ---------------------------------------------------------------------------

typedef __attribute__((ext_vector_type(16))) __bf16 v16bf;
typedef __attribute__((ext_vector_type(8)))  float  v8f;

#define N_B   2
#define N_T   2048
#define N_D   2048
#define N_H   16
#define N_DH  128
#define N_BT  4096
#define RMS_EPS 1e-6f
#define LOG_ROPE_BASE 9.210340371976184f /* ln(10000) */

__device__ inline v8f zero8() {
  v8f z;
#pragma unroll
  for (int i = 0; i < 8; ++i) z[i] = 0.f;
  return z;
}

__device__ inline v8f wmma_bf16(v16bf a, v16bf b, v8f c) {
  return __builtin_amdgcn_wmma_f32_16x16x32_bf16(
      false, a, false, b, (short)0, c, false, false);
}

// A-matrix fragment (16x32 bf16) from row-major LDS: elem (r,k) at S[(row0+r)*ld + k0+k]
__device__ inline v16bf ld_frag_rm(const __bf16* S, int ld, int row0, int k0) {
  const int lane = threadIdx.x & 31;
  const int r = lane & 15, half = lane >> 4;
  const __bf16* p = S + (size_t)(row0 + r) * ld + k0 + half * 8;
  v16bf f;
#pragma unroll
  for (int j = 0; j < 8; ++j) {
    const int off = (j < 4) ? 2 * j : 2 * j + 8;
    f[2 * j] = p[off];
    f[2 * j + 1] = p[off + 1];
  }
  return f;
}

// B-matrix fragment where logical B[k][n] = S[(n0+n)*ld + k0+k]  (Q @ K^T case)
__device__ inline v16bf ld_frag_nT(const __bf16* S, int ld, int n0, int k0) {
  const int lane = threadIdx.x & 31;
  const int n = lane & 15, half = lane >> 4;
  const __bf16* p = S + (size_t)(n0 + n) * ld + k0 + half * 8;
  v16bf f;
#pragma unroll
  for (int j = 0; j < 8; ++j) {
    const int off = (j < 4) ? 2 * j : 2 * j + 8;
    f[2 * j] = p[off];
    f[2 * j + 1] = p[off + 1];
  }
  return f;
}

// A-matrix fragment from f32 LDS with on-the-fly bf16 conversion (P tile)
__device__ inline v16bf ld_frag_f32(const float* S, int ld, int row0, int k0) {
  const int lane = threadIdx.x & 31;
  const int r = lane & 15, half = lane >> 4;
  const float* p = S + (size_t)(row0 + r) * ld + k0 + half * 8;
  v16bf f;
#pragma unroll
  for (int j = 0; j < 8; ++j) {
    const int off = (j < 4) ? 2 * j : 2 * j + 8;
    f[2 * j] = (__bf16)p[off];
    f[2 * j + 1] = (__bf16)p[off + 1];
  }
  return f;
}

// B fragment from fragment-swizzled LDS: one contiguous 32B read per lane.
// LDS layout: [block][lane][16 elems], block index supplied by caller.
__device__ inline v16bf ld_frag_swz(const __bf16* S, int block) {
  const int lane = threadIdx.x & 31;
  return *(const v16bf*)(S + ((size_t)block << 9) + lane * 16);
}

// Contiguous 32KB tile copy global->LDS via Tensor Data Mover (fallback: uint4).
__device__ inline void tile_copy_32kb(__bf16* dst_lds, const __bf16* src_g) {
#if defined(__gfx1250__) && __has_builtin(__builtin_amdgcn_tensor_load_to_lds) && \
    __has_builtin(__builtin_amdgcn_s_wait_tensorcnt)
  if (threadIdx.x < 32) {  // single wave issues the descriptor (TDM ignores EXEC)
    typedef uint32_t u32x4 __attribute__((ext_vector_type(4)));
    typedef int i32x8 __attribute__((ext_vector_type(8)));
    typedef int i32x4 __attribute__((ext_vector_type(4)));
    const uint64_t ga = (uint64_t)(uintptr_t)src_g;
    const uint32_t lds = (uint32_t)(uintptr_t)(void*)dst_lds;  // low 32 = LDS offset
    u32x4 g0 = {1u,                                            // count=1, user D#
                lds,                                           // lds_addr
                (uint32_t)ga,                                  // global_addr lo
                (uint32_t)((ga >> 32) & 0x1ffffffu) | (2u << 30)};  // type=2
    i32x8 g1 = {(int)(1u << 16),        // data_size = 2B
                (int)(16384u << 16),    // tensor_dim0 = 16384 elems
                (int)(1u << 16),        // tensor_dim1 = 1
                (int)(16384u << 16),    // tile_dim0   = 16384
                1,                      // tile_dim1   = 1
                16384, 0, 0};           // tensor_dim0_stride
    i32x4 z4 = {0, 0, 0, 0};
    i32x8 z8 = {0, 0, 0, 0, 0, 0, 0, 0};
    __builtin_amdgcn_tensor_load_to_lds(g0, g1, z4, z4, z8, 0);
    __builtin_amdgcn_s_wait_tensorcnt(0);
  }
#else
  const uint4* s4 = (const uint4*)src_g;
  uint4* d4 = (uint4*)dst_lds;
#pragma unroll
  for (int i = 0; i < 8; ++i) d4[threadIdx.x + i * 256] = s4[threadIdx.x + i * 256];
#endif
}

// ---------------------------------------------------------------------------
// f32 -> bf16 staging (row-major, for the A operand x)
// ---------------------------------------------------------------------------
__global__ __launch_bounds__(256) void cvt_kernel(const float* __restrict__ s,
                                                  __bf16* __restrict__ d, int n) {
  int i = (blockIdx.x * 256 + threadIdx.x) * 4;
  if (i + 3 < n) {
    float4 v = *(const float4*)(s + i);
    d[i + 0] = (__bf16)v.x;
    d[i + 1] = (__bf16)v.y;
    d[i + 2] = (__bf16)v.z;
    d[i + 3] = (__bf16)v.w;
  }
}

// f32 -> bf16 with WMMA-B fragment swizzle.  src row-major [K][N].
// dst[(n/16)*(K/32)+(k/32)][lane=half*16+n%16][e], half/e from k%32.
__global__ __launch_bounds__(256) void cvt_swz_kernel(const float* __restrict__ src,
                                                      __bf16* __restrict__ dst,
                                                      int K, int N) {
  const int idx = blockIdx.x * 256 + threadIdx.x;
  if (idx >= K * N) return;
  const int k = idx / N, n = idx % N;
  const int k32 = k & 31;
  const int half = (k32 >> 3) & 1;
  const int e = (k32 & 7) + ((k32 >> 4) << 3);
  const size_t blk = (size_t)(n >> 4) * (K >> 5) + (k >> 5);
  const int lane = half * 16 + (n & 15);
  dst[(blk << 9) + lane * 16 + e] = (__bf16)src[idx];
}

// ---------------------------------------------------------------------------
// Fused QKV+gate projection. Virtual N = [0,8192): [0,6144)=W_qkv, [6144,8192)=W_gate.
// Weights arrive fragment-swizzled; V written fragment-swizzled for flash PV.
// ---------------------------------------------------------------------------
__global__ __launch_bounds__(256) void proj_kernel(
    const __bf16* __restrict__ Xb, const __bf16* __restrict__ Wqkv,
    const __bf16* __restrict__ Wgate, const float* __restrict__ bgate,
    __bf16* __restrict__ Qb, __bf16* __restrict__ Kb, __bf16* __restrict__ Vswz,
    __bf16* __restrict__ Gateb) {
  __shared__ __align__(32) __bf16 As[128 * 64];
  __shared__ __align__(32) __bf16 Bs[16 * 512];  // 16 frag-blocks (2 kb x 8 nb)
  __shared__ __align__(32) float Sout[128 * 132];

  const int tid = threadIdx.x;
  const int r0 = blockIdx.x * 128;
  const int c0v = blockIdx.y * 128;
  const __bf16* W;
  int c0;
  if (c0v < 6144) { W = Wqkv;  c0 = c0v; }
  else            { W = Wgate; c0 = c0v - 6144; }
  const int c0n = c0 >> 4;          // n-block base
  const int K32 = N_D >> 5;         // 64

  const int w = tid >> 5, lane = tid & 31;
  const int wm = w & 1, wn = w >> 1;
  const int half = lane >> 4, ln = lane & 15;

  v8f acc[4][2];
#pragma unroll
  for (int mi = 0; mi < 4; ++mi)
#pragma unroll
    for (int ni = 0; ni < 2; ++ni) acc[mi][ni] = zero8();

  for (int kk = 0; kk < N_D; kk += 64) {
    if (kk + 64 < N_D)
      __builtin_prefetch(&Xb[(size_t)(r0 + (tid >> 1)) * N_D + kk + 64], 0, 1);
#pragma unroll
    for (int i = 0; i < 4; ++i) {  // A: 128x64 = 1024 x uint4, row-major
      const int ch = tid + i * 256;
      const int row = ch >> 3, c8 = ch & 7;
      *(uint4*)&As[row * 64 + c8 * 8] =
          *(const uint4*)&Xb[(size_t)(r0 + row) * N_D + kk + c8 * 8];
    }
#pragma unroll
    for (int i = 0; i < 4; ++i) {  // B: 1024 x uint4 from swizzled weights
      const int ch = tid + i * 256;
      const int nbl = ch >> 7, kbl = (ch >> 6) & 1, off = ch & 63;
      ((uint4*)Bs)[ch] = ((const uint4*)W)[
          (((size_t)(c0n + nbl) * K32 + (kk >> 5) + kbl) << 6) + off];
    }
    __syncthreads();
#pragma unroll
    for (int ks = 0; ks < 64; ks += 32) {
      v16bf af[4], bf2[2];
#pragma unroll
      for (int mi = 0; mi < 4; ++mi) af[mi] = ld_frag_rm(As, 64, wm * 64 + mi * 16, ks);
#pragma unroll
      for (int ni = 0; ni < 2; ++ni)
        bf2[ni] = ld_frag_swz(Bs, ((wn * 2 + ni) << 1) | (ks >> 5));
#pragma unroll
      for (int mi = 0; mi < 4; ++mi)
#pragma unroll
        for (int ni = 0; ni < 2; ++ni)
          acc[mi][ni] = wmma_bf16(af[mi], bf2[ni], acc[mi][ni]);
    }
    __syncthreads();
  }

#pragma unroll
  for (int mi = 0; mi < 4; ++mi)
#pragma unroll
    for (int ni = 0; ni < 2; ++ni)
#pragma unroll
      for (int i = 0; i < 8; ++i)
        Sout[(wm * 64 + mi * 16 + i + 8 * half) * 132 + wn * 32 + ni * 16 + ln] =
            acc[mi][ni][i];
  __syncthreads();

  if (tid < 128) {
    const int row = tid;
    const int R = r0 + row, bb = R >> 11, t = R & 2047;
    float* Sr = &Sout[row * 132];
    const int sec = c0v >> 11;
    const int h = (c0v >> 7) & 15;
    if (sec <= 1) {  // q or k : RMSNorm + RoPE, pack bf16 [B,H,T,dh]
      float ss = 0.f;
#pragma unroll 8
      for (int j = 0; j < 128; ++j) { float v = Sr[j]; ss += v * v; }
      const float inv = rsqrtf(ss * (1.f / 128.f) + RMS_EPS);
      __bf16* dst = (sec == 0 ? Qb : Kb) + (((size_t)(bb * N_H + h) * N_T + t) << 7);
      for (int j = 0; j < 64; ++j) {
        const float v1 = Sr[j] * inv, v2 = Sr[j + 64] * inv;
        const float fr = __expf(-(float)(2 * j) * (LOG_ROPE_BASE / 128.f));
        const float ang = (float)t * fr;
        float sn, cs;
        __sincosf(ang, &sn, &cs);
        dst[j] = (__bf16)(v1 * cs - v2 * sn);
        dst[j + 64] = (__bf16)(v1 * sn + v2 * cs);
      }
    } else if (sec == 2) {  // v : write fragment-swizzled per (b,h)
      const int kt = t >> 7, kb = (t >> 5) & 3, k32 = t & 31;
      const int vh = (k32 >> 3) & 1;
      const int e = (k32 & 7) + ((k32 >> 4) << 3);
      __bf16* base = Vswz + ((size_t)(bb * N_H + h) << 18) +
                     (((size_t)kt * 4 + kb) << 12);  // tile base (+kb*8*512)
      for (int j = 0; j < 128; ++j) {
        const int nb = j >> 4, nl = j & 15;
        base[((size_t)nb << 9) + (vh * 16 + nl) * 16 + e] = (__bf16)Sr[j];
      }
    } else {  // gate: sigmoid(x@Wg + b), layout [B,T,D]
      const float* bg = bgate + h * 128;
      __bf16* dst = Gateb + ((size_t)(bb * N_T + t) << 11) + h * 128;
      for (int j = 0; j < 128; ++j) {
        const float g = Sr[j] + bg[j];
        dst[j] = (__bf16)(1.f / (1.f + __expf(-g)));
      }
    }
  }
}

// ---------------------------------------------------------------------------
// Flash attention per (b,h,q-tile): online softmax, causal, gate fused at end.
// Q/K/V tiles are contiguous 32KB blocks -> TDM loads; V is fragment-swizzled.
// ---------------------------------------------------------------------------
__global__ __launch_bounds__(256) void flash_kernel(
    const __bf16* __restrict__ Qb, const __bf16* __restrict__ Kb,
    const __bf16* __restrict__ Vswz, const __bf16* __restrict__ Gateb,
    __bf16* __restrict__ Obuf) {
  __shared__ __align__(32) __bf16 Qs[128 * 128];
  __shared__ __align__(32) __bf16 Ks[128 * 128];
  __shared__ __align__(32) __bf16 Vs[32 * 512];  // 32 frag-blocks (4 kb x 8 nb)
  __shared__ __align__(32) float SP[128 * 132];
  __shared__ float rowm[128], rowl[128], rowa[128];

  const int tid = threadIdx.x;
  const int qt = blockIdx.x;
  const int bh = blockIdx.y;
  const int b = bh >> 4, h = bh & 15;
  const __bf16* Qg = Qb + ((size_t)bh << 18) + ((size_t)qt << 14);
  const __bf16* Kg = Kb + ((size_t)bh << 18);
  const __bf16* Vg = Vswz + ((size_t)bh << 18);

  tile_copy_32kb(Qs, Qg);
  if (tid < 128) { rowm[tid] = -3.0e38f; rowl[tid] = 0.f; }

  const int w = tid >> 5, lane = tid & 31;
  const int wm = w & 1, wn = w >> 1;
  const int half = lane >> 4, ln = lane & 15;

  v8f Oacc[4][2];
#pragma unroll
  for (int mi = 0; mi < 4; ++mi)
#pragma unroll
    for (int ni = 0; ni < 2; ++ni) Oacc[mi][ni] = zero8();

  const float scale = 0.08838834764831845f;  // 1/sqrt(128)

  for (int kt = 0; kt <= qt; ++kt) {
    __syncthreads();  // prior iteration's fragment reads complete
    tile_copy_32kb(Ks, Kg + ((size_t)kt << 14));
    tile_copy_32kb(Vs, Vg + ((size_t)kt << 14));
    __syncthreads();

    // S = Q K^T  (B fragment read transposed from K tile)
    v8f Sacc[4][2];
#pragma unroll
    for (int mi = 0; mi < 4; ++mi)
#pragma unroll
      for (int ni = 0; ni < 2; ++ni) Sacc[mi][ni] = zero8();
#pragma unroll
    for (int ks = 0; ks < 128; ks += 32) {
      v16bf af[4], bf2[2];
#pragma unroll
      for (int mi = 0; mi < 4; ++mi) af[mi] = ld_frag_rm(Qs, 128, wm * 64 + mi * 16, ks);
#pragma unroll
      for (int ni = 0; ni < 2; ++ni) bf2[ni] = ld_frag_nT(Ks, 128, wn * 32 + ni * 16, ks);
#pragma unroll
      for (int mi = 0; mi < 4; ++mi)
#pragma unroll
        for (int ni = 0; ni < 2; ++ni)
          Sacc[mi][ni] = wmma_bf16(af[mi], bf2[ni], Sacc[mi][ni]);
    }
#pragma unroll
    for (int mi = 0; mi < 4; ++mi)
#pragma unroll
      for (int ni = 0; ni < 2; ++ni)
#pragma unroll
        for (int i = 0; i < 8; ++i)
          SP[(wm * 64 + mi * 16 + i + 8 * half) * 132 + wn * 32 + ni * 16 + ln] =
              Sacc[mi][ni][i] * scale;
    __syncthreads();

    if (tid < 128) {  // online-softmax row update
      const int row = tid;
      const int lim = (kt == qt) ? (row + 1) : 128;  // causal
      float* Sr = &SP[row * 132];
      float m = rowm[row];
      for (int j = 0; j < lim; ++j) m = fmaxf(m, Sr[j]);
      const float a = __expf(rowm[row] - m);
      float l = rowl[row] * a;
      for (int j = 0; j < 128; ++j) {
        const float p = (j < lim) ? __expf(Sr[j] - m) : 0.f;
        Sr[j] = p;
        l += p;
      }
      rowm[row] = m;
      rowl[row] = l;
      rowa[row] = a;
    }
    __syncthreads();

    // rescale accumulator by alpha(row), then O += P @ V
#pragma unroll
    for (int mi = 0; mi < 4; ++mi) {
      float ar[8];
#pragma unroll
      for (int i = 0; i < 8; ++i) ar[i] = rowa[wm * 64 + mi * 16 + i + 8 * half];
#pragma unroll
      for (int ni = 0; ni < 2; ++ni)
#pragma unroll
        for (int i = 0; i < 8; ++i) Oacc[mi][ni][i] *= ar[i];
    }
#pragma unroll
    for (int ks = 0; ks < 128; ks += 32) {
      v16bf pf[4], vf[2];
#pragma unroll
      for (int mi = 0; mi < 4; ++mi) pf[mi] = ld_frag_f32(SP, 132, wm * 64 + mi * 16, ks);
#pragma unroll
      for (int ni = 0; ni < 2; ++ni)
        vf[ni] = ld_frag_swz(Vs, (ks >> 5) * 8 + (wn * 2 + ni));
#pragma unroll
      for (int mi = 0; mi < 4; ++mi)
#pragma unroll
        for (int ni = 0; ni < 2; ++ni)
          Oacc[mi][ni] = wmma_bf16(pf[mi], vf[ni], Oacc[mi][ni]);
    }
  }

  __syncthreads();
#pragma unroll
  for (int mi = 0; mi < 4; ++mi)
#pragma unroll
    for (int ni = 0; ni < 2; ++ni)
#pragma unroll
      for (int i = 0; i < 8; ++i)
        SP[(wm * 64 + mi * 16 + i + 8 * half) * 132 + wn * 32 + ni * 16 + ln] =
            Oacc[mi][ni][i];
  __syncthreads();

  if (tid < 128) {  // finalize: /l, *gate, pack bf16 into [B,T,D]
    const int row = tid;
    const int t = qt * 128 + row;
    const float invl = 1.f / rowl[row];
    const __bf16* gr = Gateb + ((size_t)(b * N_T + t)) * N_D + h * N_DH;
    __bf16* orow = Obuf + ((size_t)(b * N_T + t)) * N_D + h * N_DH;
    for (int j = 0; j < 128; ++j)
      orow[j] = (__bf16)(SP[row * 132 + j] * invl * (float)gr[j]);
  }
}

// ---------------------------------------------------------------------------
// Output projection: Obuf[4096x2048] @ Wout(swizzled) -> f32 d_out
// ---------------------------------------------------------------------------
__global__ __launch_bounds__(256) void outproj_kernel(
    const __bf16* __restrict__ Ob, const __bf16* __restrict__ Wout,
    float* __restrict__ out) {
  __shared__ __align__(32) __bf16 As[128 * 64];
  __shared__ __align__(32) __bf16 Bs[16 * 512];

  const int tid = threadIdx.x;
  const int r0 = blockIdx.x * 128;
  const int c0 = blockIdx.y * 128;
  const int c0n = c0 >> 4;
  const int K32 = N_D >> 5;
  const int w = tid >> 5, lane = tid & 31;
  const int wm = w & 1, wn = w >> 1;
  const int half = lane >> 4, ln = lane & 15;

  v8f acc[4][2];
#pragma unroll
  for (int mi = 0; mi < 4; ++mi)
#pragma unroll
    for (int ni = 0; ni < 2; ++ni) acc[mi][ni] = zero8();

  for (int kk = 0; kk < N_D; kk += 64) {
    if (kk + 64 < N_D)
      __builtin_prefetch(&Ob[(size_t)(r0 + (tid >> 1)) * N_D + kk + 64], 0, 1);
#pragma unroll
    for (int i = 0; i < 4; ++i) {
      const int ch = tid + i * 256;
      const int row = ch >> 3, c8 = ch & 7;
      *(uint4*)&As[row * 64 + c8 * 8] =
          *(const uint4*)&Ob[(size_t)(r0 + row) * N_D + kk + c8 * 8];
    }
#pragma unroll
    for (int i = 0; i < 4; ++i) {
      const int ch = tid + i * 256;
      const int nbl = ch >> 7, kbl = (ch >> 6) & 1, off = ch & 63;
      ((uint4*)Bs)[ch] = ((const uint4*)Wout)[
          (((size_t)(c0n + nbl) * K32 + (kk >> 5) + kbl) << 6) + off];
    }
    __syncthreads();
#pragma unroll
    for (int ks = 0; ks < 64; ks += 32) {
      v16bf af[4], bf2[2];
#pragma unroll
      for (int mi = 0; mi < 4; ++mi) af[mi] = ld_frag_rm(As, 64, wm * 64 + mi * 16, ks);
#pragma unroll
      for (int ni = 0; ni < 2; ++ni)
        bf2[ni] = ld_frag_swz(Bs, ((wn * 2 + ni) << 1) | (ks >> 5));
#pragma unroll
      for (int mi = 0; mi < 4; ++mi)
#pragma unroll
        for (int ni = 0; ni < 2; ++ni)
          acc[mi][ni] = wmma_bf16(af[mi], bf2[ni], acc[mi][ni]);
    }
    __syncthreads();
  }

  const int base_r = r0 + wm * 64;
  const int base_c = c0 + wn * 32;
#pragma unroll
  for (int mi = 0; mi < 4; ++mi)
#pragma unroll
    for (int ni = 0; ni < 2; ++ni)
#pragma unroll
      for (int i = 0; i < 8; ++i)
        out[(size_t)(base_r + mi * 16 + i + 8 * half) * N_D + base_c + ni * 16 + ln] =
            acc[mi][ni][i];
}

// ---------------------------------------------------------------------------
extern "C" void kernel_launch(void* const* d_in, const int* in_sizes, int n_in,
                              void* d_out, int out_size, void* d_ws, size_t ws_size,
                              hipStream_t stream) {
  const float* x      = (const float*)d_in[0];
  const float* W_qkv  = (const float*)d_in[1];
  const float* W_out  = (const float*)d_in[2];
  const float* W_gate = (const float*)d_in[3];
  const float* b_gate = (const float*)d_in[4];
  // d_in[5] (mask) unused: causal structure applied analytically.

  char* ws = (char*)d_ws;
  __bf16* Xb     = (__bf16*)(ws + 0);           //  16.8 MB  [4096,2048] row-major
  __bf16* Wqkvb  = (__bf16*)(ws + 16777216);    //  25.2 MB  swizzled
  __bf16* Wgateb = (__bf16*)(ws + 41943040);    //   8.4 MB  swizzled
  __bf16* Woutb  = (__bf16*)(ws + 50331648);    //   8.4 MB  swizzled
  __bf16* Qb     = (__bf16*)(ws + 58720256);    //  16.8 MB  [B,H,T,dh]
  __bf16* Kb     = (__bf16*)(ws + 75497472);    //  16.8 MB  [B,H,T,dh]
  __bf16* Vswz   = (__bf16*)(ws + 92274688);    //  16.8 MB  frag-swizzled per (b,h)
  __bf16* Gateb  = (__bf16*)(ws + 109051904);   //  16.8 MB  [B,T,D]
  __bf16* Obuf   = (__bf16*)(ws + 125829120);   //  16.8 MB  [B,T,D]

  cvt_kernel<<<(N_BT * N_D) / 1024, 256, 0, stream>>>(x, Xb, N_BT * N_D);
  cvt_swz_kernel<<<(N_D * 3 * N_D) / 256, 256, 0, stream>>>(W_qkv, Wqkvb, N_D, 3 * N_D);
  cvt_swz_kernel<<<(N_D * N_D) / 256, 256, 0, stream>>>(W_gate, Wgateb, N_D, N_D);
  cvt_swz_kernel<<<(N_D * N_D) / 256, 256, 0, stream>>>(W_out, Woutb, N_D, N_D);

  proj_kernel<<<dim3(N_BT / 128, 8192 / 128), 256, 0, stream>>>(
      Xb, Wqkvb, Wgateb, b_gate, Qb, Kb, Vswz, Gateb);

  flash_kernel<<<dim3(N_T / 128, N_B * N_H), 256, 0, stream>>>(Qb, Kb, Vswz, Gateb, Obuf);

  outproj_kernel<<<dim3(N_BT / 128, N_D / 128), 256, 0, stream>>>(
      Obuf, Woutb, (float*)d_out);
}
